// PersonaGrammarDecoder_52647709114847
// MI455X (gfx1250) — matrix operand order repaired and checked by reading, not compiled, not run on hardware
//
#include <hip/hip_runtime.h>

// ---------------- problem constants ----------------
constexpr int B  = 64;
constexpr int S  = 64;
constexpr int T  = 20;
constexpr int E  = 512;
constexpr int H  = 1024;
constexpr int V  = 32000;
constexpr int KC = 100;   // attention key dim
constexpr int LC = 50;    // persona context length
constexpr int WC = 8;     // words per context item
constexpr int GH = 4 * H;     // 4096 gate width
constexpr int EH = E + H;     // 1536
constexpr int H2 = 2 * H;     // 2048

typedef __attribute__((ext_vector_type(16))) __bf16 v16bf;
typedef __attribute__((ext_vector_type(8)))  float  v8f;

// float -> bf16 bits, round-to-nearest-even
__device__ __forceinline__ unsigned short f2bf(float f) {
  union { float f; unsigned int u; } x; x.f = f;
  unsigned int u = x.u;
  unsigned int r = (u + 0x7FFFu + ((u >> 16) & 1u)) >> 16;
  return (unsigned short)r;
}

__device__ __forceinline__ float sigm(float x) { return 1.0f / (1.0f + __expf(-x)); }

// ---------------- elementwise / gather kernels ----------------
__global__ void k_f32_to_bf16(const float* __restrict__ x, unsigned short* __restrict__ y, int n) {
  int i = blockIdx.x * blockDim.x + threadIdx.x;
  if (i < n) y[i] = f2bf(x[i]);
}

__global__ void k_gather_rows(const unsigned short* __restrict__ emb, const int* __restrict__ ids,
                              int nrows, unsigned short* __restrict__ out) {
  int i = blockIdx.x * blockDim.x + threadIdx.x;
  int total = nrows * (E / 8);
  if (i >= total) return;
  int r  = i / (E / 8);
  int cg = (i % (E / 8)) * 8;
  const uint4* src = (const uint4*)(emb + (size_t)ids[r] * E + cg);
  uint4* dst       = (uint4*)(out + (size_t)r * E + cg);
  *dst = *src;
}

__global__ void k_ctx_mean(const float* __restrict__ emb, const int* __restrict__ ctc,
                           float* __restrict__ outf, unsigned short* __restrict__ outb) {
  int i = blockIdx.x * blockDim.x + threadIdx.x;
  int total = B * LC * E;
  if (i >= total) return;
  int e  = i % E;
  int bl = i / E;
  const int* ids = ctc + (size_t)bl * WC;
  float s = 0.0f;
  #pragma unroll
  for (int w = 0; w < WC; ++w) s += emb[(size_t)ids[w] * E + e];
  s *= (1.0f / (float)WC);
  outf[i] = s;
  outb[i] = f2bf(s);
}

// ---------------- WMMA fragment helpers (ISA 7.12.2 wave32 layouts) ----------------
// A 16x32 bf16: VGPR v holds K = {2v (v<4) | 16+2(v-4)} + 8*half, row M = lane&15
__device__ __forceinline__ v16bf frag_a(const unsigned short* __restrict__ row, int kk, int half) {
  union { unsigned short u[16]; v16bf v; } f;
  #pragma unroll
  for (int p = 0; p < 8; ++p) {
    int ka = kk + ((p < 4) ? (2 * p) : (16 + 2 * (p - 4))) + 8 * half;
    f.u[2 * p]     = row[ka];
    f.u[2 * p + 1] = row[ka + 1];
  }
  return f.v;
}
// B 32x16 bf16: VGPR v holds K = {2v (v<4) | 16+2(v-4)} + 8*half, col N = lane&15 (row = weight row n)
__device__ __forceinline__ v16bf frag_b(const unsigned short* __restrict__ row, int kk, int half) {
  union { unsigned short u[16]; v16bf v; } f;
  #pragma unroll
  for (int p = 0; p < 8; ++p) {
    int kb = kk + ((p < 4) ? (2 * p) : (16 + 2 * (p - 4))) + 8 * half;
    f.u[2 * p]     = row[kb];
    f.u[2 * p + 1] = row[kb + 1];
  }
  return f.v;
}

// issue one async 16B global->LDS copy (ASYNCcnt-tracked, no VGPR staging)
__device__ __forceinline__ void async_b128(unsigned lds_addr, const void* gaddr) {
  asm volatile("global_load_async_to_lds_b128 %0, %1, off"
               :: "v"(lds_addr), "v"((unsigned long long)gaddr)
               : "memory");
}
__device__ __forceinline__ void wait_async0() {
  asm volatile("s_wait_asynccnt 0x0" ::: "memory");
}

// ---------------- WMMA bf16 GEMM: C[M,N] = act(A[M,K] @ W[N,K]^T + bias) ----------------
// Requirements (hold for every call in this model): M % 64 == 0, K % 64 == 0.
// act: 0=none 1=relu 2=tanh; optional bf16 mirror Cbf.
__global__ __launch_bounds__(256)
void gemm_bf16(const unsigned short* __restrict__ A, int lda,
               const unsigned short* __restrict__ Bw, int ldb,
               const float* __restrict__ bias,
               float* __restrict__ C, int ldc,
               unsigned short* __restrict__ Cbf,
               int M, int N, int K, int act) {
  constexpr int BM = 64, BN = 128, BK = 64, LDP = BK + 8;  // 72 shorts = 144B row (16B aligned)
  __shared__ unsigned short As[BM][LDP];
  __shared__ unsigned short Bs[BN][LDP];

  const int tid  = threadIdx.x;
  const int wave = tid >> 5;
  const int lane = tid & 31;
  const int wm   = wave & 1;    // M half: 0..1  -> 32 rows each
  const int wn   = wave >> 1;   // N quarter: 0..3 -> 32 cols each
  const int half = lane >> 4;
  const int l16  = lane & 15;

  const int m0 = blockIdx.y * BM;
  const int n0 = blockIdx.x * BN;
  const bool fullN = (n0 + BN) <= N;

  v8f acc[2][2] = {};

  for (int k0 = 0; k0 < K; k0 += BK) {
    if (fullN) {
      // ---- async fast path: global -> LDS without VGPR staging ----
      #pragma unroll
      for (int i = 0; i < 2; ++i) {            // A: 64x64 = 512 x b128
        int idx = tid + i * 256;
        int r = idx >> 3, cg = (idx & 7) * 8;
        async_b128((unsigned)(unsigned long long)&As[r][cg],
                   A + (size_t)(m0 + r) * lda + k0 + cg);
      }
      #pragma unroll
      for (int i = 0; i < 4; ++i) {            // B: 128x64 = 1024 x b128
        int idx = tid + i * 256;
        int r = idx >> 3, cg = (idx & 7) * 8;
        async_b128((unsigned)(unsigned long long)&Bs[r][cg],
                   Bw + (size_t)(n0 + r) * ldb + k0 + cg);
      }
      wait_async0();
      __syncthreads();
    } else {
      // ---- guarded path (ragged N tiles): zero-fill OOB weight rows ----
      #pragma unroll
      for (int i = 0; i < 2; ++i) {
        int idx = tid + i * 256;
        int r = idx >> 3, cg = (idx & 7) * 8;
        *(uint4*)(&As[r][cg]) = *(const uint4*)(A + (size_t)(m0 + r) * lda + k0 + cg);
      }
      #pragma unroll
      for (int i = 0; i < 4; ++i) {
        int idx = tid + i * 256;
        int r = idx >> 3, cg = (idx & 7) * 8;
        uint4 val = make_uint4(0u, 0u, 0u, 0u);
        if (n0 + r < N) val = *(const uint4*)(Bw + (size_t)(n0 + r) * ldb + k0 + cg);
        *(uint4*)(&Bs[r][cg]) = val;
      }
      __syncthreads();
    }

    // ---- 2 K-chunks x (2x2) WMMAs: 8 matrix ops per barrier pair ----
    #pragma unroll
    for (int kk = 0; kk < BK; kk += 32) {
      v16bf a0 = frag_a(&As[wm * 32 + l16][0],      kk, half);
      v16bf a1 = frag_a(&As[wm * 32 + 16 + l16][0], kk, half);
      v16bf b0 = frag_b(&Bs[wn * 32 + l16][0],      kk, half);
      v16bf b1 = frag_b(&Bs[wn * 32 + 16 + l16][0], kk, half);
      acc[0][0] = __builtin_amdgcn_wmma_f32_16x16x32_bf16(false, a0, false, b0, (short)0, acc[0][0], false, false);
      acc[0][1] = __builtin_amdgcn_wmma_f32_16x16x32_bf16(false, a0, false, b1, (short)0, acc[0][1], false, false);
      acc[1][0] = __builtin_amdgcn_wmma_f32_16x16x32_bf16(false, a1, false, b0, (short)0, acc[1][0], false, false);
      acc[1][1] = __builtin_amdgcn_wmma_f32_16x16x32_bf16(false, a1, false, b1, (short)0, acc[1][1], false, false);
    }
    __syncthreads();
  }

  // ---- epilogue: D layout — VGPR r: M = r + 8*half, N = lane&15 ----
  #pragma unroll
  for (int mi = 0; mi < 2; ++mi) {
    #pragma unroll
    for (int ni = 0; ni < 2; ++ni) {
      int gn = n0 + wn * 32 + ni * 16 + l16;
      if (gn < N) {
        #pragma unroll
        for (int r = 0; r < 8; ++r) {
          int gm = m0 + wm * 32 + mi * 16 + r + 8 * half;
          float v = acc[mi][ni][r];
          if (bias) v += bias[gn];
          if (act == 1) v = fmaxf(v, 0.0f);
          else if (act == 2) v = tanhf(v);
          C[(size_t)gm * ldc + gn] = v;
          if (Cbf) Cbf[(size_t)gm * ldc + gn] = f2bf(v);
        }
      }
    }
  }
}

// ---------------- fused LSTM gates (PyTorch order i,f,g,o) ----------------
__global__ void k_lstm_gates(const float* __restrict__ xg, long xg_row_stride,
                             const float* __restrict__ hg,
                             const float* __restrict__ bih, const float* __restrict__ bhh,
                             float* __restrict__ h, float* __restrict__ c,
                             unsigned short* __restrict__ h_bf,
                             const int* __restrict__ lengths, int t,
                             float* __restrict__ h_last, float* __restrict__ c_last) {
  int idx = blockIdx.x * blockDim.x + threadIdx.x;
  if (idx >= B * H) return;
  int b = idx / H, j = idx % H;
  const float* xr = xg + (size_t)b * xg_row_stride;
  const float* hr = hg + (size_t)b * GH;
  float gi = xr[j]          + hr[j]          + bih[j]          + bhh[j];
  float gf = xr[H + j]      + hr[H + j]      + bih[H + j]      + bhh[H + j];
  float gg = xr[2 * H + j]  + hr[2 * H + j]  + bih[2 * H + j]  + bhh[2 * H + j];
  float go = xr[3 * H + j]  + hr[3 * H + j]  + bih[3 * H + j]  + bhh[3 * H + j];
  float cc = sigm(gf) * c[idx] + sigm(gi) * tanhf(gg);
  float hh = sigm(go) * tanhf(cc);
  c[idx] = cc;
  h[idx] = hh;
  h_bf[idx] = f2bf(hh);
  if (lengths && t == lengths[b] - 1) { h_last[idx] = hh; c_last[idx] = cc; }
}

// ---------------- masked-softmax attention + context (tiny: VALU) ----------------
__global__ void k_attn(const float* __restrict__ qkey,   // [B*L, KC]
                       const float* __restrict__ qval,   // [B*L, H]
                       const float* __restrict__ akey,   // [B, KC]
                       const int* __restrict__ lens,
                       float* __restrict__ ctx) {        // [B, H]
  int b   = blockIdx.x;
  int tid = threadIdx.x;
  __shared__ float e[LC];
  __shared__ float inv_s;
  if (tid < LC) {
    float s = 0.0f;
    const float* qr = qkey + (size_t)(b * LC + tid) * KC;
    const float* ar = akey + (size_t)b * KC;
    for (int k = 0; k < KC; ++k) s += qr[k] * ar[k];
    e[tid] = (tid < lens[b]) ? s : -1e9f;
  }
  __syncthreads();
  if (tid == 0) {
    float m = -3.0e38f;
    for (int l = 0; l < LC; ++l) m = fmaxf(m, e[l]);
    float s = 0.0f;
    for (int l = 0; l < LC; ++l) { e[l] = __expf(e[l] - m); s += e[l]; }
    inv_s = 1.0f / s;
  }
  __syncthreads();
  float inv = inv_s;
  for (int hh = tid; hh < H; hh += blockDim.x) {
    float s = 0.0f;
    for (int l = 0; l < LC; ++l)
      s += e[l] * qval[(size_t)(b * LC + l) * H + hh];
    ctx[(size_t)b * H + hh] = s * inv;
  }
}

__global__ void k_concat_xcat(const unsigned short* __restrict__ ansb, int t,
                              const float* __restrict__ ctx, unsigned short* __restrict__ out) {
  int i = blockIdx.x * blockDim.x + threadIdx.x;
  if (i >= B * EH) return;
  int b = i / EH, j = i % EH;
  out[i] = (j < E) ? ansb[(size_t)(b * T + t) * E + j] : f2bf(ctx[(size_t)b * H + (j - E)]);
}

__global__ void k_concat_hc(const float* __restrict__ h, const float* __restrict__ ctx,
                            unsigned short* __restrict__ out) {
  int i = blockIdx.x * blockDim.x + threadIdx.x;
  if (i >= B * H2) return;
  int b = i / H2, j = i % H2;
  float v = (j < H) ? h[(size_t)b * H + j] : ctx[(size_t)b * H + (j - H)];
  out[i] = f2bf(v);
}

// ---------------- host launch ----------------
static inline void launch_gemm(hipStream_t s, const unsigned short* A, int lda,
                               const unsigned short* Bw, int ldb, const float* bias,
                               float* C, int ldc, unsigned short* Cbf,
                               int M, int N, int K, int act) {
  dim3 g((N + 127) / 128, (M + 63) / 64);
  gemm_bf16<<<g, 256, 0, s>>>(A, lda, Bw, ldb, bias, C, ldc, Cbf, M, N, K, act);
}

static inline void launch_conv(hipStream_t s, const float* x, unsigned short* y, int n) {
  k_f32_to_bf16<<<(n + 255) / 256, 256, 0, s>>>(x, y, n);
}

extern "C" void kernel_launch(void* const* d_in, const int* in_sizes, int n_in,
                              void* d_out, int out_size, void* d_ws, size_t ws_size,
                              hipStream_t stream) {
  // ---- inputs (setup_inputs order) ----
  const int*   src_seqs    = (const int*)d_in[0];
  const int*   src_lengths = (const int*)d_in[1];
  const int*   trg_seqs    = (const int*)d_in[2];
  const int*   ctc_seqs    = (const int*)d_in[3];
  const int*   ctc_lengths = (const int*)d_in[4];
  const float* embed_W = (const float*)d_in[5];
  const float* enc_Wih = (const float*)d_in[6];
  const float* enc_Whh = (const float*)d_in[7];
  const float* enc_bih = (const float*)d_in[8];
  const float* enc_bhh = (const float*)d_in[9];
  const float* dec_Wih = (const float*)d_in[10];
  const float* dec_Whh = (const float*)d_in[11];
  const float* dec_bih = (const float*)d_in[12];
  const float* dec_bhh = (const float*)d_in[13];
  const float* qk_W = (const float*)d_in[14];
  const float* qk_b = (const float*)d_in[15];
  const float* qv_W = (const float*)d_in[16];
  const float* qv_b = (const float*)d_in[17];
  const float* ak_W = (const float*)d_in[18];
  const float* ak_b = (const float*)d_in[19];
  const float* out_W = (const float*)d_in[20];
  const float* out_b = (const float*)d_in[21];
  const float* wd_b  = (const float*)d_in[22];
  const float* hf1_W = (const float*)d_in[23];
  const float* hf1_b = (const float*)d_in[24];
  const float* hf2_W = (const float*)d_in[25];
  const float* hf2_b = (const float*)d_in[26];
  const float* cf1_W = (const float*)d_in[27];
  const float* cf1_b = (const float*)d_in[28];
  const float* cf2_W = (const float*)d_in[29];
  const float* cf2_b = (const float*)d_in[30];

  float* out = (float*)d_out;  // [B, T-1, V]

  // ---- workspace carve-up ----
  char* base = (char*)d_ws;
  size_t off = 0;
  auto alloc = [&](size_t bytes) -> char* {
    off = (off + 255) & ~(size_t)255;
    char* p = base + off;
    off += bytes;
    return p;
  };
  auto aB = [&](size_t elems) { return (unsigned short*)alloc(elems * 2); };
  auto aF = [&](size_t elems) { return (float*)alloc(elems * 4); };

  // bf16 weights
  unsigned short* Wemb  = aB((size_t)V * E);
  unsigned short* WihE  = aB((size_t)GH * E);
  unsigned short* WhhE  = aB((size_t)GH * H);
  unsigned short* WihD  = aB((size_t)GH * EH);
  unsigned short* WhhD  = aB((size_t)GH * H);
  unsigned short* Wqk   = aB((size_t)KC * E);
  unsigned short* Wqv   = aB((size_t)H * E);
  unsigned short* Wak   = aB((size_t)KC * H);
  unsigned short* Wout  = aB((size_t)E * H2);
  unsigned short* Whf1  = aB((size_t)H2 * H);
  unsigned short* Whf2  = aB((size_t)H * H2);
  unsigned short* Wcf1  = aB((size_t)H2 * H);
  unsigned short* Wcf2  = aB((size_t)H * H2);
  // activations
  unsigned short* Asrc  = aB((size_t)B * S * E);
  unsigned short* Ansb  = aB((size_t)B * T * E);
  float*          Xg    = aF((size_t)B * S * GH);
  float*          Ctxf  = aF((size_t)B * LC * E);
  unsigned short* Ctxb  = aB((size_t)B * LC * E);
  float*          Qkey  = aF((size_t)B * LC * KC);
  float*          Qval  = aF((size_t)B * LC * H);
  float*          hbuf  = aF((size_t)B * H);
  float*          cbuf  = aF((size_t)B * H);
  float*          hlast = aF((size_t)B * H);
  float*          clast = aF((size_t)B * H);
  unsigned short* hbf   = aB((size_t)B * H);
  unsigned short* hlbf  = aB((size_t)B * H);
  unsigned short* clbf  = aB((size_t)B * H);
  float*          t1f   = aF((size_t)B * H2);
  unsigned short* t1b   = aB((size_t)B * H2);
  float*          hg    = aF((size_t)B * GH);
  float*          xgd   = aF((size_t)B * GH);
  float*          akey  = aF((size_t)B * KC);
  float*          actx  = aF((size_t)B * H);
  unsigned short* xcat  = aB((size_t)B * EH);
  unsigned short* hcb   = aB((size_t)B * H2);
  float*          oxf   = aF((size_t)B * E);
  unsigned short* oxb   = aB((size_t)B * E);
  (void)ws_size; (void)n_in; (void)in_sizes; (void)out_size;
  (void)Ctxf;

  // ---- 1. weight conversion fp32 -> bf16 ----
  launch_conv(stream, embed_W, Wemb, V * E);
  launch_conv(stream, enc_Wih, WihE, GH * E);
  launch_conv(stream, enc_Whh, WhhE, GH * H);
  launch_conv(stream, dec_Wih, WihD, GH * EH);
  launch_conv(stream, dec_Whh, WhhD, GH * H);
  launch_conv(stream, qk_W,  Wqk,  KC * E);
  launch_conv(stream, qv_W,  Wqv,  H * E);
  launch_conv(stream, ak_W,  Wak,  KC * H);
  launch_conv(stream, out_W, Wout, E * H2);
  launch_conv(stream, hf1_W, Whf1, H2 * H);
  launch_conv(stream, hf2_W, Whf2, H * H2);
  launch_conv(stream, cf1_W, Wcf1, H2 * H);
  launch_conv(stream, cf2_W, Wcf2, H * H2);

  // ---- 2. embedding gathers / persona context mean ----
  {
    int n = B * S * (E / 8);
    k_gather_rows<<<(n + 255) / 256, 256, 0, stream>>>(Wemb, src_seqs, B * S, Asrc);
    n = B * T * (E / 8);
    k_gather_rows<<<(n + 255) / 256, 256, 0, stream>>>(Wemb, trg_seqs, B * T, Ansb);
    n = B * LC * E;
    k_ctx_mean<<<(n + 255) / 256, 256, 0, stream>>>(embed_W, ctc_seqs, Ctxf, Ctxb);
  }

  // ---- 3. encoder input GEMM hoisted over time: Xg[b*S+s,:] = src_embed @ Wih^T ----
  launch_gemm(stream, Asrc, E, WihE, E, nullptr, Xg, GH, nullptr, B * S, GH, E, 0);

  // ---- 4. persona projections ----
  launch_gemm(stream, Ctxb, E, Wqk, E, qk_b, Qkey, KC, nullptr, B * LC, KC, E, 2 /*tanh*/);
  launch_gemm(stream, Ctxb, E, Wqv, E, qv_b, Qval, H,  nullptr, B * LC, H,  E, 0);

  // ---- 5. encoder recurrence ----
  hipMemsetAsync(hbuf, 0, (size_t)B * H * 4, stream);
  hipMemsetAsync(cbuf, 0, (size_t)B * H * 4, stream);
  hipMemsetAsync(hbf,  0, (size_t)B * H * 2, stream);
  for (int t = 0; t < S; ++t) {
    launch_gemm(stream, hbf, H, WhhE, H, nullptr, hg, GH, nullptr, B, GH, H, 0);
    k_lstm_gates<<<(B * H + 255) / 256, 256, 0, stream>>>(
        Xg + (size_t)t * GH, (long)S * GH, hg, enc_bih, enc_bhh,
        hbuf, cbuf, hbf, src_lengths, t, hlast, clast);
  }

  // ---- 6. init_hidden MLPs ----
  launch_conv(stream, hlast, hlbf, B * H);
  launch_conv(stream, clast, clbf, B * H);
  launch_gemm(stream, hlbf, H,  Whf1, H,  hf1_b, t1f, H2, t1b, B, H2, H,  1 /*relu*/);
  launch_gemm(stream, t1b,  H2, Whf2, H2, hf2_b, hbuf, H, hbf, B, H,  H2, 0);
  launch_gemm(stream, clbf, H,  Wcf1, H,  cf1_b, t1f, H2, t1b, B, H2, H,  1 /*relu*/);
  launch_gemm(stream, t1b,  H2, Wcf2, H2, cf2_b, cbuf, H, nullptr, B, H, H2, 0);

  // ---- 7. decoder: attention + LSTM + tied-vocab projection ----
  for (int t = 0; t < T - 1; ++t) {
    launch_gemm(stream, hbf, H, Wak, H, ak_b, akey, KC, nullptr, B, KC, H, 2 /*tanh*/);
    k_attn<<<B, 256, 0, stream>>>(Qkey, Qval, akey, ctc_lengths, actx);
    k_concat_xcat<<<(B * EH + 255) / 256, 256, 0, stream>>>(Ansb, t, actx, xcat);
    launch_gemm(stream, xcat, EH, WihD, EH, nullptr, xgd, GH, nullptr, B, GH, EH, 0);
    launch_gemm(stream, hbf,  H,  WhhD, H,  nullptr, hg,  GH, nullptr, B, GH, H,  0);
    k_lstm_gates<<<(B * H + 255) / 256, 256, 0, stream>>>(
        xgd, (long)GH, hg, dec_bih, dec_bhh, hbuf, cbuf, hbf,
        nullptr, 0, nullptr, nullptr);
    k_concat_hc<<<(B * H2 + 255) / 256, 256, 0, stream>>>(hbuf, actx, hcb);
    launch_gemm(stream, hcb, H2, Wout, H2, out_b, oxf, E, oxb, B, E, H2, 0);
    launch_gemm(stream, oxb, E, Wemb, E, wd_b, out + (size_t)t * V, (T - 1) * V,
                nullptr, B, V, E, 0);
  }
}
